// TarMAC_3298534883830
// MI455X (gfx1250) — matrix operand order repaired and checked by reading, not compiled, not run on hardware
//
#include <hip/hip_runtime.h>

// ---------------------------------------------------------------------------
// TarMAC batched attention for gfx1250 (MI455X), wave32 + v_wmma_f32_16x16x32_f16
// E=128 envs, NB=256 agents, DQ=DK=DV=256, D=768.
// Per block (256 thr = 8 waves): half an env (128 q rows), wave = 16-row q tile.
//   P1: stage K (qkv+noise -> f16) to LDS            (coalesced float4, unroll 4)
//   P2: Q in A-layout regs; 128x v_wmma for Q K^T    (B tiles software-pipelined)
//   P3: mask + softmax fully in registers (shfl width-16 reductions)
//   P3b: stage full P tile (f16) to LDS once -> frees 128 score VGPRs
//   P4: stream V in 32-key transposed chunks through LDS; v_wmma for P V
//   P5: coalesced f32 stores
// ---------------------------------------------------------------------------

#define EENV 128
#define NBA  256
#define DQv  256
#define DKv  256
#define DVv  256
#define DD   768

typedef __attribute__((ext_vector_type(16))) _Float16 v16h;
typedef __attribute__((ext_vector_type(8)))  _Float16 v8h;
typedef __attribute__((ext_vector_type(8)))  float    v8f;

// LDS layout (dynamic):
//   [0, 135168)   : K as f16, 256 rows x KSTR halves          (phases 1-2)
//                   aliased after softmax barrier by:
//                     [0, 20480)        Vt: 256 d-rows x VSTR halves (32 keys)
//                     [20480, 88064)    P: 8 waves x 16 rows x PFSTR halves
//   [135168, +2048): bot positions, 256 x {x,y} f32
#define KSTR  264                // 256 + 8 pad halves; row = 528 B (16B aligned, banks +4/lane)
#define VSTR  40                 // 32 + 8 pad halves; row = 80 B (16B aligned, banks +20/row)
#define PFSTR 264                // full P row stride (same padding properties as KSTR)
#define VT_HALVES   (256 * VSTR)             // 10240 halves = 20480 B
#define PST_OFF     VT_HALVES
#define KLDS_HALVES (256 * KSTR)             // 67584 halves = 135168 B
#define POS_OFF_B   (KLDS_HALVES * 2)
#define SMEM_BYTES  (POS_OFF_B + NBA * 2 * 4)    // 137216 B (< 160KB CU-mode limit)

static __device__ __forceinline__ v16h cat8(v8h lo, v8h hi) {
    return __builtin_shufflevector(lo, hi, 0,1,2,3,4,5,6,7,8,9,10,11,12,13,14,15);
}

__global__ __launch_bounds__(256)
void tarmac_attn_kernel(const float* __restrict__ qkv,
                        const int*   __restrict__ mask,     // 0/1 audibility
                        const float* __restrict__ bot_pos,
                        const float* __restrict__ noise,
                        const float* __restrict__ s_ptr,
                        float*       __restrict__ out)
{
    extern __shared__ char smem[];
    _Float16* Klds = (_Float16*)smem;
    _Float16* Vt   = (_Float16*)smem;                 // aliases Klds (phase 4)
    _Float16* Pst  = ((_Float16*)smem) + PST_OFF;     // aliases Klds (phase 3b/4)
    float*    pos  = (float*)(smem + POS_OFF_B);

    const int tid    = threadIdx.x;
    const int lane   = tid & 31;
    const int wave   = tid >> 5;
    const int lane16 = lane & 15;
    const int laneHi = lane >> 4;

    const int env = blockIdx.x >> 1;                  // 2 blocks per env
    const int wq0 = (blockIdx.x & 1) * 128 + wave * 16;  // this wave's first q row

    const float scale = fabsf(s_ptr[0]) + 1.0f;

    // ---------------- Phase 0: stage agent positions ----------------
    if (tid < NBA) {
        pos[2 * tid]     = bot_pos[((size_t)env * NBA + tid) * 2];
        pos[2 * tid + 1] = bot_pos[((size_t)env * NBA + tid) * 2 + 1];
    }

    // ---------------- Phase 1: stage K (qkv+noise -> f16) into LDS ----------------
    {
        const float4* qb = (const float4*)(qkv   + (size_t)env * NBA * DD);
        const float4* nb = (const float4*)(noise + (size_t)env * NBA * DD);
        #pragma unroll 4
        for (int it = 0; it < 64; ++it) {             // 256*64 float4 = 256x256 floats
            int flat = tid + it * 256;
            int key  = flat >> 6;
            int d4   = flat & 63;
            int gi   = key * (DD / 4) + (DQv / 4) + d4;
            float4 a = qb[gi], b = nb[gi];
            _Float16* dst = Klds + key * KSTR + d4 * 4;
            dst[0] = (_Float16)(a.x + b.x);
            dst[1] = (_Float16)(a.y + b.y);
            dst[2] = (_Float16)(a.z + b.z);
            dst[3] = (_Float16)(a.w + b.w);
        }
    }
    __syncthreads();

    // ---------------- Phase 2: load Q into A-layout regs, scores = Q K^T ---------
    // A layout (16-bit, 16x32): lane m = lane16; half h -> k = (h/8)*16 + laneHi*8 + h%8
    v16h qa[8];
    {
        const size_t qg = (size_t)env * NBA + wq0 + lane16;
        const float4* q4 = (const float4*)(qkv   + qg * DD);
        const float4* n4 = (const float4*)(noise + qg * DD);
        #pragma unroll
        for (int kc = 0; kc < 8; ++kc) {
            int b0 = kc * 8 + laneHi * 2;             // float4 index of d = kc*32 + laneHi*8
            float4 x0 = q4[b0],     y0 = n4[b0];
            float4 x1 = q4[b0 + 1], y1 = n4[b0 + 1];
            float4 x2 = q4[b0 + 4], y2 = n4[b0 + 4];  // +16 floats
            float4 x3 = q4[b0 + 5], y3 = n4[b0 + 5];
            v16h a;
            a[0]=(_Float16)((x0.x+y0.x)*scale); a[1]=(_Float16)((x0.y+y0.y)*scale);
            a[2]=(_Float16)((x0.z+y0.z)*scale); a[3]=(_Float16)((x0.w+y0.w)*scale);
            a[4]=(_Float16)((x1.x+y1.x)*scale); a[5]=(_Float16)((x1.y+y1.y)*scale);
            a[6]=(_Float16)((x1.z+y1.z)*scale); a[7]=(_Float16)((x1.w+y1.w)*scale);
            a[8]=(_Float16)((x2.x+y2.x)*scale); a[9]=(_Float16)((x2.y+y2.y)*scale);
            a[10]=(_Float16)((x2.z+y2.z)*scale); a[11]=(_Float16)((x2.w+y2.w)*scale);
            a[12]=(_Float16)((x3.x+y3.x)*scale); a[13]=(_Float16)((x3.y+y3.y)*scale);
            a[14]=(_Float16)((x3.z+y3.z)*scale); a[15]=(_Float16)((x3.w+y3.w)*scale);
            qa[kc] = a;
        }
    }

    // B layout (16-bit, 32x16): lane n = lane16; half h -> k = (h/8)*16 + laneHi*8 + h%8
    // Flattened 128-tile stream (i = nt*8 + kc), software-pipelined 2 tiles deep.
    v8f sc[16];
    {
        const _Float16* kr = Klds + lane16 * KSTR + laneHi * 8;
        #define KTILE(i) (kr + ((i) >> 3) * (16 * KSTR) + ((i) & 7) * 32)
        v8h blo[3], bhi[3];
        blo[0] = *(const v8h*)KTILE(0); bhi[0] = *(const v8h*)(KTILE(0) + 16);
        blo[1] = *(const v8h*)KTILE(1); bhi[1] = *(const v8h*)(KTILE(1) + 16);
        v8f acc = {};
        #pragma unroll
        for (int i = 0; i < 128; ++i) {
            const int cur = i % 3;
            if (i + 2 < 128) {
                const int nxt = (i + 2) % 3;
                blo[nxt] = *(const v8h*)KTILE(i + 2);
                bhi[nxt] = *(const v8h*)(KTILE(i + 2) + 16);
            }
            acc = __builtin_amdgcn_wmma_f32_16x16x32_f16(
                      false, qa[i & 7], false, cat8(blo[cur], bhi[cur]),
                      (short)0, acc, false, false);
            if ((i & 7) == 7) { sc[i >> 3] = acc; acc = (v8f){}; }
        }
        #undef KTILE
    }

    // ---------------- Phase 3: mask + softmax in registers ----------------
    // C layout: accum vgpr r, lane -> element (m = laneHi*8 + r, n = lane16)
    float qpx[8], qpy[8], mx[8], sum[8];
    int   anyr[8];
    #pragma unroll
    for (int r = 0; r < 8; ++r) {
        int mr = wq0 + laneHi * 8 + r;
        qpx[r] = pos[2 * mr]; qpy[r] = pos[2 * mr + 1];
        mx[r] = -3.0e38f; sum[r] = 0.0f; anyr[r] = 0;
    }
    const int* mrow = mask + ((size_t)env * NBA + wq0 + laneHi * 8) * NBA;
    #pragma unroll
    for (int nt = 0; nt < 16; ++nt) {
        int key = nt * 16 + lane16;
        float kx = pos[2 * key], ky = pos[2 * key + 1];
        #pragma unroll
        for (int r = 0; r < 8; ++r) {
            int aud = mrow[r * NBA + key];
            float dx = qpx[r] - kx, dy = qpy[r] - ky;
            aud = (aud != 0) && (dx * dx + dy * dy < 225.0f);   // COM_RANGE^2
            float v = sc[nt][r] * 0.0625f + (aud ? 0.0f : -1000000.0f);
            sc[nt][r] = v;
            anyr[r] |= aud;
            mx[r] = fmaxf(mx[r], v);
        }
    }
    #pragma unroll
    for (int r = 0; r < 8; ++r) {
        #pragma unroll
        for (int off = 1; off < 16; off <<= 1) {                // reduce across 16-lane half
            mx[r]  = fmaxf(mx[r], __shfl_xor(mx[r], off, 32));
            anyr[r] |= __shfl_xor(anyr[r], off, 32);
        }
    }
    #pragma unroll
    for (int nt = 0; nt < 16; ++nt)
        #pragma unroll
        for (int r = 0; r < 8; ++r) {
            float p = __expf(sc[nt][r] - mx[r]);
            sc[nt][r] = p;
            sum[r] += p;
        }
    #pragma unroll
    for (int r = 0; r < 8; ++r) {
        #pragma unroll
        for (int off = 1; off < 16; off <<= 1)
            sum[r] += __shfl_xor(sum[r], off, 32);
        float rin = 1.0f / sum[r];
        rin = anyr[r] ? rin : 0.0f;       // fold "row has any audible" into normalization
        #pragma unroll
        for (int nt = 0; nt < 16; ++nt) sc[nt][r] *= rin;
    }

    // ---------------- Phase 3b: stage full P tile (f16) to LDS once --------------
    // Frees all 128 score VGPRs before the output accumulators come alive.
    __syncthreads();                       // all waves done reading Klds; safe to alias
    _Float16* pw = Pst + wave * 16 * PFSTR;
    #pragma unroll
    for (int nt = 0; nt < 16; ++nt)
        #pragma unroll
        for (int r = 0; r < 8; ++r)
            pw[(laneHi * 8 + r) * PFSTR + nt * 16 + lane16] = (_Float16)sc[nt][r];

    // ---------------- Phase 4: out = P V  (V streamed through LDS, transposed) ---
    v8f oacc[16];
    #pragma unroll
    for (int dt = 0; dt < 16; ++dt) oacc[dt] = (v8f){};

    for (int kc = 0; kc < 8; ++kc) {
        __syncthreads();   // previous Vt chunk fully consumed
        {   // stage transposed V chunk: keys [kc*32, kc*32+32), all 256 d
            const float* qb = qkv   + ((size_t)env * NBA + kc * 32) * DD + DQv + DKv;
            const float* nb = noise + ((size_t)env * NBA + kc * 32) * DD + DQv + DKv;
            #pragma unroll 4
            for (int i = 0; i < 32; ++i) {
                int flat = tid + i * 256;
                int kl = flat >> 8, d = flat & 255;
                float vv = qb[(size_t)kl * DD + d] + nb[(size_t)kl * DD + d];
                Vt[d * VSTR + kl] = (_Float16)vv;
            }
        }
        __syncthreads();

        // reload this wave's P slice in A-layout (row = lane16, k = kc*32 + kA)
        const _Float16* pr = pw + lane16 * PFSTR + kc * 32 + laneHi * 8;
        v16h pa = cat8(*(const v8h*)pr, *(const v8h*)(pr + 16));

        // 16 d-tiles, software-pipelined 2 tiles deep
        const _Float16* vr = Vt + lane16 * VSTR + laneHi * 8;
        v8h vlo[3], vhi[3];
        vlo[0] = *(const v8h*)vr;                 vhi[0] = *(const v8h*)(vr + 16);
        vlo[1] = *(const v8h*)(vr + 16 * VSTR);   vhi[1] = *(const v8h*)(vr + 16 * VSTR + 16);
        #pragma unroll
        for (int dt = 0; dt < 16; ++dt) {
            const int cur = dt % 3;
            if (dt + 2 < 16) {
                const int nxt = (dt + 2) % 3;
                const _Float16* vp = vr + (dt + 2) * 16 * VSTR;
                vlo[nxt] = *(const v8h*)vp;
                vhi[nxt] = *(const v8h*)(vp + 16);
            }
            oacc[dt] = __builtin_amdgcn_wmma_f32_16x16x32_f16(
                           false, pa, false, cat8(vlo[cur], vhi[cur]),
                           (short)0, oacc[dt], false, false);
        }
    }

    // ---------------- Phase 5: coalesced f32 stores ----------------
    float* ob = out + ((size_t)env * NBA + wq0 + laneHi * 8) * DVv + lane16;
    #pragma unroll
    for (int dt = 0; dt < 16; ++dt)
        #pragma unroll
        for (int r = 0; r < 8; ++r)
            ob[(size_t)r * DVv + dt * 16] = oacc[dt][r];
}

extern "C" void kernel_launch(void* const* d_in, const int* in_sizes, int n_in,
                              void* d_out, int out_size, void* d_ws, size_t ws_size,
                              hipStream_t stream) {
    const float* qkv     = (const float*)d_in[0];
    const int*   mask    = (const int*)d_in[1];   // bool -> int per harness convention
    const float* bot_pos = (const float*)d_in[2];
    const float* noise   = (const float*)d_in[3];
    const float* s       = (const float*)d_in[4];
    float*       out     = (float*)d_out;

    dim3 grid(EENV * 2);     // 2 workgroups per env (128 q rows each)
    dim3 block(256);         // 8 waves of 32
    tarmac_attn_kernel<<<grid, block, SMEM_BYTES, stream>>>(qkv, mask, bot_pos, noise, s, out);
}
